// MultiHeadLatentAttention_32607391711699
// MI455X (gfx1250) — matrix operand order repaired and checked by reading, not compiled, not run on hardware
//
#include <hip/hip_runtime.h>
#include <math.h>
#include <stdint.h>

typedef __attribute__((ext_vector_type(16))) _Float16 v16h;
typedef __attribute__((ext_vector_type(8)))  float    v8f;

#define D_IN 2048
#define D_OUT 2048
#define N_HEAD 16
#define N_KV 4
#define HEAD_DIM 128
#define KV_DIM 64
#define LATENT 1024
#define SEQ 2048
#define BATCH 2
#define GROUP 4
#define QKV_COLS (N_HEAD * KV_DIM)   /* 1024 */
#define KV_COLS  (N_KV * KV_DIM)     /* 256  */
#define ROWS     (BATCH * SEQ)       /* 4096 */
#define NEG_INF (-1.0e30f)

__device__ __forceinline__ v8f wmma_f16(v16h a, v16h b, v8f c) {
  return __builtin_amdgcn_wmma_f32_16x16x32_f16(false, a, false, b, (short)0, c,
                                                false, false);
}

// AS(3) pointers on amdgcn are byte offsets into the workgroup LDS allocation,
// which is exactly what GLOBAL_LOAD_ASYNC_TO_LDS expects in its LDS-dest VGPR.
__device__ __forceinline__ unsigned lds_offset(const void* p) {
  return (unsigned)(unsigned long long)(__attribute__((address_space(3))) const void*)p;
}

// Async DMA: 16 bytes per lane, global -> LDS, tracked by ASYNCcnt.
__device__ __forceinline__ void async_copy_b128(unsigned lds_dst, const void* gsrc) {
  asm volatile("global_load_async_to_lds_b128 %0, %1, off"
               :
               : "v"(lds_dst), "v"((unsigned long long)(uintptr_t)gsrc)
               : "memory");
}

__device__ __forceinline__ void wait_async0() {
  asm volatile("s_wait_asynccnt 0" ::: "memory");
}

// ---------------------------------------------------------------------------
// Fold kernels: Wq_eff = per-head Wq @ Wq2kv ; Wo_eff = Wkv2h @ per-head Wo
// ---------------------------------------------------------------------------
__global__ void fold_wq_kernel(const float* __restrict__ Wq,
                               const float* __restrict__ Wq2kv,
                               float* __restrict__ Wq_eff) {
  int col = blockIdx.x * blockDim.x + threadIdx.x;  // 0..1023  (h*64+e)
  int row = blockIdx.y;                             // 0..2047  (d_in)
  int h = col >> 6, e = col & 63;
  const float* wq = Wq + (size_t)row * D_OUT + h * HEAD_DIM;
  float s = 0.f;
  for (int c = 0; c < HEAD_DIM; ++c) s += wq[c] * Wq2kv[c * KV_DIM + e];
  Wq_eff[(size_t)row * QKV_COLS + col] = s;
}

__global__ void fold_bq_kernel(const float* __restrict__ bq,
                               const float* __restrict__ Wq2kv,
                               float* __restrict__ bq_eff) {
  int col = blockIdx.x * blockDim.x + threadIdx.x;  // 0..1023
  int h = col >> 6, e = col & 63;
  float s = 0.f;
  for (int c = 0; c < HEAD_DIM; ++c) s += bq[h * HEAD_DIM + c] * Wq2kv[c * KV_DIM + e];
  bq_eff[col] = s;
}

__global__ void fold_wo_kernel(const float* __restrict__ Wkv2h,
                               const float* __restrict__ Wo,
                               float* __restrict__ Wo_eff) {
  int n = blockIdx.x * blockDim.x + threadIdx.x;  // 0..2047
  int row = blockIdx.y;                           // 0..1023  (h*64+e)
  int h = row >> 6, e = row & 63;
  float s = 0.f;
  for (int c = 0; c < HEAD_DIM; ++c)
    s += Wkv2h[e * HEAD_DIM + c] * Wo[(size_t)(h * HEAD_DIM + c) * D_OUT + n];
  Wo_eff[(size_t)row * D_OUT + n] = s;
}

__global__ void fold_bo_kernel(const float* __restrict__ bkv2h,
                               const float* __restrict__ Wo,
                               const float* __restrict__ bo,
                               float* __restrict__ bo_eff) {
  int n = blockIdx.x * blockDim.x + threadIdx.x;  // 0..2047
  float s = bo[n];
  for (int j = 0; j < D_OUT; ++j)
    s += bkv2h[j & (HEAD_DIM - 1)] * Wo[(size_t)j * D_OUT + n];
  bo_eff[n] = s;
}

// ---------------------------------------------------------------------------
// fp32-in / f16-WMMA / fp32-out GEMM: C = act(A[M,K] @ B[K,N] + bias)
// 256 threads = 8 waves; block tile 128x128, K-step 32.
// Double-buffered async global->LDS staging (raw fp32), converted to f16 tiles.
// A tile stored row-major [row][k]; B tile stored col-major [col][k] so every
// fragment read is 2x contiguous ds_load_b128.
// Wave grid 4(M) x 2(N); each wave: 2x4 tiles of v_wmma_f32_16x16x32_f16.
// ---------------------------------------------------------------------------
#define BM 128
#define BN 128
#define BK 32
#define AP 40    /* As row stride (halves): 80 B, keeps 16 B alignment */
#define BPC 40   /* Bs col stride (halves) */

template <int ACT>
__global__ __launch_bounds__(256) void gemm_wmma_kernel(
    const float* __restrict__ A, const float* __restrict__ B,
    const float* __restrict__ bias, float* __restrict__ C,
    int M, int N, int K) {
  __shared__ float    Araw[2][BM * BK];   // 2 x 16 KB raw fp32
  __shared__ float    Braw[2][BK * BN];   // 2 x 16 KB raw fp32
  __shared__ _Float16 As[BM * AP];        // [row][k]
  __shared__ _Float16 Bs[BN * BPC];       // [col][k]  (transposed at convert)

  int tid  = threadIdx.x;
  int bm   = blockIdx.y * BM;
  int bn   = blockIdx.x * BN;
  int wave = tid >> 5;
  int lane = tid & 31;
  int half = lane >> 4;
  int lr   = lane & 15;
  int wm   = (wave & 3) * 32;   // 4 waves along M
  int wn   = (wave >> 2) * 64;  // 2 waves along N

  v8f zero = {};
  v8f acc[2][4];
#pragma unroll
  for (int i = 0; i < 2; ++i)
#pragma unroll
    for (int j = 0; j < 4; ++j) acc[i][j] = zero;

  int arow = tid >> 1, aseg = (tid & 1) * 16;  // A tile: 128 rows x 32 (16 f/thread)
  int brow = tid >> 3, bseg = (tid & 7) * 16;  // B tile: 32 rows x 128 (16 f/thread)

  // Issue async copies of one K-tile into buffer `buf` (4x b128 each for A, B).
  auto issue_tile = [&](int k0, int buf) {
    const float* ag = A + (size_t)(bm + arow) * K + k0 + aseg;
    const float* bg = B + (size_t)(k0 + brow) * N + bn + bseg;
#pragma unroll
    for (int j = 0; j < 4; ++j)
      async_copy_b128(lds_offset(&Araw[buf][arow * BK + aseg + 4 * j]), ag + 4 * j);
#pragma unroll
    for (int j = 0; j < 4; ++j)
      async_copy_b128(lds_offset(&Braw[buf][brow * BN + bseg + 4 * j]), bg + 4 * j);
  };

  issue_tile(0, 0);
  int nk = K / BK;
  for (int t = 0; t < nk; ++t) {
    int cur = t & 1;
    wait_async0();       // this wave's async copies (tile t) have landed
    __syncthreads();     // ... and everyone else's too; prev compute done

    // Convert raw fp32 -> f16 WMMA tiles (each thread converts what it fetched).
    // A stays row-major; B is transposed to col-major here (write-once strided).
    {
      const float* ar = &Araw[cur][arow * BK + aseg];
#pragma unroll
      for (int j = 0; j < 16; ++j) As[arow * AP + aseg + j] = (_Float16)ar[j];
      const float* br = &Braw[cur][brow * BN + bseg];
#pragma unroll
      for (int j = 0; j < 16; ++j) Bs[(bseg + j) * BPC + brow] = (_Float16)br[j];
    }
    if (t + 1 < nk) issue_tile((t + 1) * BK, cur ^ 1);  // overlap DMA with compute
    __syncthreads();

    // Preload ALL fragments first (contiguous b128 reads), then the WMMA batch.
    // A frag: lane row lane%16; K = 16*(lane/16) + i, contiguous
    v16h afr[2];
    int kb0 = half * 8;
#pragma unroll
    for (int mi = 0; mi < 2; ++mi) {
      const _Float16* ap = &As[(wm + mi * 16 + lr) * AP];
#pragma unroll
      for (int i = 0; i < 8; ++i) {
        afr[mi][i]     = ap[2 * kb0 + i];
        afr[mi][i + 8] = ap[2 * kb0 + 8 + i];
      }
    }
    // B frag: lane col lane%16; K = 16*(lane/16) + i, contiguous in col-major Bs
    v16h bfr[4];
#pragma unroll
    for (int ni = 0; ni < 4; ++ni) {
      const _Float16* bp = &Bs[(wn + ni * 16 + lr) * BPC + half * 16];
#pragma unroll
      for (int i = 0; i < 16; ++i) bfr[ni][i] = bp[i];
    }
#pragma unroll
    for (int ni = 0; ni < 4; ++ni)
#pragma unroll
      for (int mi = 0; mi < 2; ++mi)
        acc[mi][ni] = wmma_f16(afr[mi], bfr[ni], acc[mi][ni]);
  }

  // Epilogue: C layout per ISA (VGPR r -> row r + 8*(lane/16), col lane%16)
#pragma unroll
  for (int mi = 0; mi < 2; ++mi) {
#pragma unroll
    for (int ni = 0; ni < 4; ++ni) {
      int col  = bn + wn + ni * 16 + lr;
      float bb = bias[col];
#pragma unroll
      for (int r = 0; r < 8; ++r) {
        int row = bm + wm + mi * 16 + half * 8 + r;
        float v = acc[mi][ni][r] + bb;
        if (ACT == 1) v = v / (1.f + __expf(-v));  // SiLU
        C[(size_t)row * N + col] = v;
      }
    }
  }
}

// ---------------------------------------------------------------------------
// Causal flash attention over q_kv [b,t,h*64+d], k/v [b,t,kvh*64+d].
// Block = (qtile of 128 rows) x (b,h). 8 waves; each wave owns 16 q rows.
// K/V tiles of 32 keys double-buffered via async global->LDS, converted to f16.
// K tile stored [key][d] (frag reads contiguous in d); V tile stored [d][key]
// (frag reads contiguous in key). S via WMMA; online softmax; O += P@V via WMMA.
// ---------------------------------------------------------------------------
__global__ __launch_bounds__(256) void flash_attn_kernel(
    const float* __restrict__ qkv, const float* __restrict__ kbuf,
    const float* __restrict__ vbuf, float* __restrict__ ctx) {
  __shared__ float    Kraw[2][32 * 64];   // 2 x 8 KB raw fp32
  __shared__ float    Vraw[2][32 * 64];
  __shared__ _Float16 Kt[32 * 72];        // [key][d], 144 B row stride
  __shared__ _Float16 Vt[64 * 40];        // [d][key], 80 B row stride
  __shared__ _Float16 Pw[8 * 16 * 40];    // per-wave P scratch (16 q x 32 keys)

  int tid  = threadIdx.x;
  int wave = tid >> 5, lane = tid & 31;
  int half = lane >> 4, lr = lane & 15;
  int bh  = blockIdx.y;
  int b   = bh >> 4;   // / N_HEAD
  int h   = bh & 15;
  int kvh = h >> 2;    // grouped KV head (repeat by 4)
  int q0  = blockIdx.x * 128;
  int qw0 = q0 + wave * 16;
  int kb0 = half * 8;

  // Q fragments (two 16x32 A tiles covering d=0..63), pre-scaled by 1/sqrt(64)
  v16h aQ[2];
  {
    const float* qp = qkv + (size_t)(b * SEQ + qw0 + lr) * QKV_COLS + h * KV_DIM;
#pragma unroll
    for (int c = 0; c < 2; ++c) {
#pragma unroll
      for (int i = 0; i < 8; ++i) {
        aQ[c][i]     = (_Float16)(qp[c * 32 + 2 * kb0 + i] * 0.125f);
        aQ[c][i + 8] = (_Float16)(qp[c * 32 + 2 * kb0 + 8 + i] * 0.125f);
      }
    }
  }

  v8f zero = {};
  v8f o[4];
#pragma unroll
  for (int j = 0; j < 4; ++j) o[j] = zero;
  float mrow[8], lrow[8];
#pragma unroll
  for (int r = 0; r < 8; ++r) { mrow[r] = NEG_INF; lrow[r] = 0.f; }

  int skey = tid >> 3;           // 0..31
  int sseg = (tid & 7) * 8;      // 0..56 (8 floats = 2x b128 per thread)
  int ntiles = (q0 + 128) / 32;  // causal: keys 0 .. q0+127
  _Float16* pw = Pw + wave * 16 * 40;

  auto issue_kv = [&](int kk, int buf) {
    const float* kp = kbuf + (size_t)(b * SEQ + kk + skey) * KV_COLS + kvh * KV_DIM + sseg;
    const float* vp = vbuf + (size_t)(b * SEQ + kk + skey) * KV_COLS + kvh * KV_DIM + sseg;
#pragma unroll
    for (int j = 0; j < 2; ++j) {
      async_copy_b128(lds_offset(&Kraw[buf][skey * 64 + sseg + 4 * j]), kp + 4 * j);
      async_copy_b128(lds_offset(&Vraw[buf][skey * 64 + sseg + 4 * j]), vp + 4 * j);
    }
  };

  issue_kv(0, 0);
  for (int t = 0; t < ntiles; ++t) {
    int kk = t * 32;
    int cur = t & 1;
    wait_async0();
    __syncthreads();

    // Convert raw fp32 K/V -> f16 tiles; V is transposed to [d][key] here.
    {
      const float* kr = &Kraw[cur][skey * 64 + sseg];
      const float* vr = &Vraw[cur][skey * 64 + sseg];
#pragma unroll
      for (int j = 0; j < 8; ++j) {
        Kt[skey * 72 + sseg + j]  = (_Float16)kr[j];
        Vt[(sseg + j) * 40 + skey] = (_Float16)vr[j];
      }
    }
    if (t + 1 < ntiles) issue_kv(kk + 32, cur ^ 1);  // overlap DMA with compute
    __syncthreads();

    // Preload all K fragments (4 = 2 key-subtiles x 2 d-chunks), then 4 WMMAs.
    v16h bk[2][2];
#pragma unroll
    for (int sub = 0; sub < 2; ++sub)
#pragma unroll
      for (int c = 0; c < 2; ++c) {
        const _Float16* kp2 = &Kt[(sub * 16 + lr) * 72 + c * 32 + half * 16];
#pragma unroll
        for (int i = 0; i < 16; ++i) bk[sub][c][i] = kp2[i];
      }
    v8f s[2];
#pragma unroll
    for (int sub = 0; sub < 2; ++sub) {
      s[sub] = zero;
#pragma unroll
      for (int c = 0; c < 2; ++c) s[sub] = wmma_f16(aQ[c], bk[sub][c], s[sub]);
    }

    // Online softmax update (rows striped across half-waves per C layout)
    int key0 = kk + lr, key1 = kk + 16 + lr;
#pragma unroll
    for (int r = 0; r < 8; ++r) {
      int qrow = qw0 + half * 8 + r;
      float v0 = (key0 <= qrow) ? s[0][r] : NEG_INF;
      float v1 = (key1 <= qrow) ? s[1][r] : NEG_INF;
      float mx = fmaxf(v0, v1);
#pragma unroll
      for (int off = 1; off < 16; off <<= 1) mx = fmaxf(mx, __shfl_xor(mx, off, 16));
      float mnew  = fmaxf(mrow[r], mx);
      float alpha = __expf(mrow[r] - mnew);
      float p0 = __expf(v0 - mnew);
      float p1 = __expf(v1 - mnew);
      float ps = p0 + p1;
#pragma unroll
      for (int off = 1; off < 16; off <<= 1) ps += __shfl_xor(ps, off, 16);
      lrow[r] = lrow[r] * alpha + ps;
      mrow[r] = mnew;
#pragma unroll
      for (int j = 0; j < 4; ++j) o[j][r] *= alpha;
      // spill P in C layout to per-wave LDS for the transpose to A layout
      pw[(half * 8 + r) * 40 + lr]      = (_Float16)p0;
      pw[(half * 8 + r) * 40 + 16 + lr] = (_Float16)p1;
    }
    // wave-private LDS RAW: wait LDS stores, then scheduling barrier
    asm volatile("s_wait_dscnt 0" ::: "memory");
    __builtin_amdgcn_wave_barrier();

    // P as 16x32 A fragment
    v16h aP;
#pragma unroll
    for (int i = 0; i < 8; ++i) {
      aP[i]     = pw[lr * 40 + 2 * kb0 + i];
      aP[i + 8] = pw[lr * 40 + 2 * kb0 + 8 + i];
    }
    // Preload all V fragments (contiguous in key after transpose), then 4 WMMAs
    v16h bv[4];
#pragma unroll
    for (int j = 0; j < 4; ++j) {
      const _Float16* vp2 = &Vt[(j * 16 + lr) * 40 + half * 16];
#pragma unroll
      for (int i = 0; i < 16; ++i) bv[j][i] = vp2[i];
    }
#pragma unroll
    for (int j = 0; j < 4; ++j) o[j] = wmma_f16(aP, bv[j], o[j]);
  }

  // normalize and write ctx [b, t, h*64 + d]
#pragma unroll
  for (int j = 0; j < 4; ++j) {
#pragma unroll
    for (int r = 0; r < 8; ++r) {
      int trow  = qw0 + half * 8 + r;
      float inv = 1.f / lrow[r];
      ctx[(size_t)(b * SEQ + trow) * QKV_COLS + h * KV_DIM + j * 16 + lr] =
          o[j][r] * inv;
    }
  }
}

// ---------------------------------------------------------------------------
extern "C" void kernel_launch(void* const* d_in, const int* in_sizes, int n_in,
                              void* d_out, int out_size, void* d_ws, size_t ws_size,
                              hipStream_t stream) {
  const float* x     = (const float*)d_in[0];
  const float* Wq    = (const float*)d_in[1];
  const float* bq    = (const float*)d_in[2];
  const float* Wl    = (const float*)d_in[3];
  const float* bl    = (const float*)d_in[4];
  const float* Wk    = (const float*)d_in[5];
  const float* bk    = (const float*)d_in[6];
  const float* Wv    = (const float*)d_in[7];
  const float* bv    = (const float*)d_in[8];
  const float* Wq2kv = (const float*)d_in[9];
  const float* Wkv2h = (const float*)d_in[10];
  const float* bkv2h = (const float*)d_in[11];
  const float* Wo    = (const float*)d_in[12];
  const float* bo    = (const float*)d_in[13];
  float* out = (float*)d_out;

  float* p      = (float*)d_ws;
  float* Wq_eff = p; p += (size_t)D_IN * QKV_COLS;   // 2048x1024
  float* bq_eff = p; p += QKV_COLS;
  float* Wo_eff = p; p += (size_t)QKV_COLS * D_OUT;  // 1024x2048
  float* bo_eff = p; p += D_OUT;
  float* qkv    = p; p += (size_t)ROWS * QKV_COLS;   // [b*t, 1024]
  float* lat    = p; p += (size_t)ROWS * LATENT;     // reused as ctx
  float* kbf    = p; p += (size_t)ROWS * KV_COLS;
  float* vbf    = p; p += (size_t)ROWS * KV_COLS;
  float* ctx    = lat;  // lat is dead after K/V projections

  fold_wq_kernel<<<dim3(QKV_COLS / 256, D_IN), 256, 0, stream>>>(Wq, Wq2kv, Wq_eff);
  fold_bq_kernel<<<dim3(QKV_COLS / 256), 256, 0, stream>>>(bq, Wq2kv, bq_eff);
  fold_wo_kernel<<<dim3(D_OUT / 256, QKV_COLS), 256, 0, stream>>>(Wkv2h, Wo, Wo_eff);
  fold_bo_kernel<<<dim3(D_OUT / 256), 256, 0, stream>>>(bkv2h, Wo, bo, bo_eff);

  // q_kv = x @ Wq_eff + bq_eff
  gemm_wmma_kernel<0><<<dim3(QKV_COLS / BN, ROWS / BM), 256, 0, stream>>>(
      x, Wq_eff, bq_eff, qkv, ROWS, QKV_COLS, D_IN);
  // lat = silu(x @ Wl + bl)
  gemm_wmma_kernel<1><<<dim3(LATENT / BN, ROWS / BM), 256, 0, stream>>>(
      x, Wl, bl, lat, ROWS, LATENT, D_IN);
  // k = lat @ Wk + bk ; v = lat @ Wv + bv
  gemm_wmma_kernel<0><<<dim3(KV_COLS / BN, ROWS / BM), 256, 0, stream>>>(
      lat, Wk, bk, kbf, ROWS, KV_COLS, LATENT);
  gemm_wmma_kernel<0><<<dim3(KV_COLS / BN, ROWS / BM), 256, 0, stream>>>(
      lat, Wv, bv, vbf, ROWS, KV_COLS, LATENT);

  // causal grouped attention -> ctx (overwrites lat buffer)
  flash_attn_kernel<<<dim3(SEQ / 128, BATCH * N_HEAD), 256, 0, stream>>>(
      qkv, kbf, vbf, ctx);

  // out = ctx @ Wo_eff + bo_eff
  gemm_wmma_kernel<0><<<dim3(D_OUT / BN, ROWS / BM), 256, 0, stream>>>(
      ctx, Wo_eff, bo_eff, out, ROWS, D_OUT, QKV_COLS);
}
